// SLM_69054484185383
// MI455X (gfx1250) — matrix operand order repaired and checked by reading, not compiled, not run on hardware
//
#include <hip/hip_runtime.h>

// ---------------- problem constants ----------------
#define B_   64
#define P_   256
#define V_   65
#define H_   6
#define D_   128
#define C_   768            // H_*D_
#define FF_  3072           // 4*C_
#define M_   (B_*P_)        // 16384 rows

typedef __bf16 bf16_t;
typedef __attribute__((ext_vector_type(16))) __bf16 v16bf;
typedef __attribute__((ext_vector_type(8)))  __bf16 v8bf;
typedef __attribute__((ext_vector_type(8)))  float  v8f;

// -------- gfx1250 async global->LDS path (probe via __has_builtin) --------
// Builtin signature (from hipcc diagnostic): arg0 = v4i32 addrspace(1)*,
// arg1 = v4i32 addrspace(3)*, then two immediate ints (offset, cpol).
typedef int v4i_gcc __attribute__((vector_size(16)));
typedef __attribute__((address_space(1))) v4i_gcc* g4ptr;
typedef __attribute__((address_space(3))) v4i_gcc* l4ptr;

#if defined(__has_builtin)
#if __has_builtin(__builtin_amdgcn_global_load_async_to_lds_b128)
#define USE_ASYNC_LDS 1
#else
#define USE_ASYNC_LDS 0
#endif
#if __has_builtin(__builtin_amdgcn_s_wait_asynccnt)
#define WAIT_ASYNC0() __builtin_amdgcn_s_wait_asynccnt(0)
#else
#define WAIT_ASYNC0() asm volatile("s_wait_asynccnt 0" ::: "memory")
#endif
#else
#define USE_ASYNC_LDS 0
#define WAIT_ASYNC0()
#endif

#if USE_ASYNC_LDS
#define ASYNC_CP16(gsrc, ldst)                                                  \
    __builtin_amdgcn_global_load_async_to_lds_b128(                             \
        (g4ptr)(gsrc), (l4ptr)(ldst), 0, 0)
#endif

// ---------------- WMMA helpers ----------------
__device__ __forceinline__ v8f wmma_bf16(v16bf a, v16bf b, v8f c) {
    return __builtin_amdgcn_wmma_f32_16x16x32_bf16(
        false, a, false, b, (short)0, c, false, false);
}

__device__ __forceinline__ v8f zero8() {
    v8f z;
#pragma unroll
    for (int i = 0; i < 8; ++i) z[i] = 0.0f;
    return z;
}

// A-matrix 16x32 bf16 fragment (ISA 7.12.2):
// lanes 0-15: row M=lane, K={0..7},{16..23}; lanes 16-31: K={8..15},{24..31}
__device__ __forceinline__ v16bf load_fragA(const bf16_t* base, int stride, int lane) {
    const int row  = lane & 15;
    const int koff = (lane >> 4) << 3;
    const bf16_t* p = base + row * stride + koff;
    v8bf lo = *reinterpret_cast<const v8bf*>(p);
    v8bf hi = *reinterpret_cast<const v8bf*>(p + 16);
    v16bf f;
#pragma unroll
    for (int i = 0; i < 8; ++i) { f[i] = lo[i]; f[i + 8] = hi[i]; }
    return f;
}

// B-matrix 32x16 bf16 fragment: lanes 0-15 carry K={0..7},{16..23}; lanes 16-31 K={8..15},{24..31}
__device__ __forceinline__ v16bf load_fragB(const bf16_t* base, int stride, int lane) {
    const int col  = lane & 15;
    const int koff = (lane >> 4) << 3;
    const bf16_t* p = base + col * stride + koff;
    v8bf lo = *reinterpret_cast<const v8bf*>(p);
    v8bf hi = *reinterpret_cast<const v8bf*>(p + 16);
    v16bf f;
#pragma unroll
    for (int i = 0; i < 8; ++i) { f[i] = lo[i]; f[i + 8] = hi[i]; }
    return f;
}

// ---------------- wave32 reductions ----------------
__device__ __forceinline__ float wave_max(float v) {
#pragma unroll
    for (int m = 16; m >= 1; m >>= 1) v = fmaxf(v, __shfl_xor(v, m, 32));
    return v;
}
__device__ __forceinline__ float wave_sum(float v) {
#pragma unroll
    for (int m = 16; m >= 1; m >>= 1) v += __shfl_xor(v, m, 32);
    return v;
}

// ---------------- elementwise kernels ----------------
__global__ __launch_bounds__(256) void embed_kernel(
    const int* __restrict__ idx, const float* __restrict__ tok,
    const float* __restrict__ pos, float* __restrict__ x)
{
    int t = blockIdx.x * 256 + threadIdx.x;
    if (t >= M_ * C_) return;
    int bp = t / C_, c = t - bp * C_;
    int p = bp & (P_ - 1);
    x[t] = tok[(size_t)idx[bp] * C_ + c] + pos[(size_t)p * C_ + c];
}

// one wave per row; 8 rows per 256-thread block
__global__ __launch_bounds__(256) void ln_kernel(
    const float* __restrict__ x, const float* __restrict__ g,
    const float* __restrict__ b, bf16_t* __restrict__ out, int M, int Cc)
{
    int wave = threadIdx.x >> 5, lane = threadIdx.x & 31;
    int row = blockIdx.x * 8 + wave;
    if (row >= M) return;
    const float* xr = x + (size_t)row * Cc;
    float s = 0.0f, s2 = 0.0f;
    for (int c = lane; c < Cc; c += 32) { float t = xr[c]; s += t; s2 += t * t; }
    s = wave_sum(s); s2 = wave_sum(s2);
    float mean = s / (float)Cc;
    float var  = s2 / (float)Cc - mean * mean;
    float inv  = rsqrtf(var + 1e-5f);
    bf16_t* orow = out + (size_t)row * Cc;
    for (int c = lane; c < Cc; c += 32)
        orow[c] = (bf16_t)((xr[c] - mean) * inv * g[c] + b[c]);
}

// src [K, Nreal] f32 -> dst [Npad, K] bf16 (zero-fill rows >= Nreal)
__global__ __launch_bounds__(256) void transpose_conv(
    const float* __restrict__ src, bf16_t* __restrict__ dst,
    int K, int Nreal, int Npad)
{
    int t = blockIdx.x * 256 + threadIdx.x;
    if (t >= Npad * K) return;
    int n = t / K, k = t - n * K;
    float v = (n < Nreal) ? src[(size_t)k * Nreal + n] : 0.0f;
    dst[t] = (bf16_t)v;
}

// src [H][C][D] f32 -> dst [n=h*D+d][C] bf16
__global__ __launch_bounds__(256) void head_transpose(
    const float* __restrict__ src, bf16_t* __restrict__ dst)
{
    int t = blockIdx.x * 256 + threadIdx.x;
    if (t >= C_ * C_) return;
    int n = t / C_, c = t - n * C_;
    int h = n / D_, d = n - h * D_;
    dst[t] = (bf16_t)src[((size_t)h * C_ + c) * D_ + d];
}

// ---------------- WMMA GEMM: C[M,N] = A[M,K] * Bt[N,K]^T ----------------
// block tile 128x128, 8 waves (4x2), wave tile 32x64 (2x4 wmma tiles).
// Double-buffered LDS; async global->LDS prefetch when available.
template<bool BIAS, bool RES, bool GELU_, bool STF, bool STB, bool NG>
__global__ __launch_bounds__(256) void gemm_kernel(
    const bf16_t* __restrict__ A, const bf16_t* __restrict__ Bt,
    const float* __restrict__ bias, const float* __restrict__ resid,
    float* __restrict__ outF, bf16_t* __restrict__ outB,
    int K, int ldo, int Nreal)
{
    __shared__ bf16_t As[2][128 * 32];
    __shared__ bf16_t Bs[2][128 * 32];

    const int tid  = threadIdx.x;
    const int lane = tid & 31;
    const int wave = tid >> 5;
    const int wr   = wave >> 1;   // 0..3
    const int wc   = wave & 1;    // 0..1
    const int m0 = blockIdx.y * 128;
    const int n0 = blockIdx.x * 128;

    const int ldRow  = tid >> 1;          // 0..127
    const int ldHalf = (tid & 1) << 4;    // 0 or 16

    const bf16_t* gA = A  + (size_t)(m0 + ldRow) * K + ldHalf;
    const bf16_t* gB = Bt + (size_t)(n0 + ldRow) * K + ldHalf;
    const int lofs = ldRow * 32 + ldHalf;

    v8f acc[2][4];
#pragma unroll
    for (int tm = 0; tm < 2; ++tm)
#pragma unroll
        for (int tn = 0; tn < 4; ++tn) acc[tm][tn] = zero8();

#if USE_ASYNC_LDS
    // ---- async double-buffered pipeline: 1 barrier / K-step ----
    ASYNC_CP16(gA,     &As[0][lofs]);
    ASYNC_CP16(gA + 8, &As[0][lofs + 8]);
    ASYNC_CP16(gB,     &Bs[0][lofs]);
    ASYNC_CP16(gB + 8, &Bs[0][lofs + 8]);
    for (int k0 = 0; k0 < K; k0 += 32) {
        const int cur = (k0 >> 5) & 1;
        WAIT_ASYNC0();          // our buf[cur] writes have landed in LDS
        __syncthreads();        // everyone's landed; prev reads of buf[cur^1] done
        if (k0 + 32 < K) {      // overlap next tile's HBM traffic with WMMAs
            const int nxt = cur ^ 1;
            ASYNC_CP16(gA + k0 + 32,     &As[nxt][lofs]);
            ASYNC_CP16(gA + k0 + 40,     &As[nxt][lofs + 8]);
            ASYNC_CP16(gB + k0 + 32,     &Bs[nxt][lofs]);
            ASYNC_CP16(gB + k0 + 40,     &Bs[nxt][lofs + 8]);
        }
        v16bf af[2], bfr[4];
#pragma unroll
        for (int tm = 0; tm < 2; ++tm)
            af[tm] = load_fragA(&As[cur][(wr * 32 + tm * 16) * 32], 32, lane);
#pragma unroll
        for (int tn = 0; tn < 4; ++tn)
            bfr[tn] = load_fragB(&Bs[cur][(wc * 64 + tn * 16) * 32], 32, lane);
#pragma unroll
        for (int tm = 0; tm < 2; ++tm)
#pragma unroll
            for (int tn = 0; tn < 4; ++tn)
                acc[tm][tn] = wmma_bf16(af[tm], bfr[tn], acc[tm][tn]);
    }
#else
    // ---- fallback: register prefetch + alternating LDS buffers ----
    v8bf a0 = *(const v8bf*)gA,       a1 = *(const v8bf*)(gA + 8);
    v8bf b0 = *(const v8bf*)gB,       b1 = *(const v8bf*)(gB + 8);
    *(v8bf*)&As[0][lofs] = a0;  *(v8bf*)&As[0][lofs + 8] = a1;
    *(v8bf*)&Bs[0][lofs] = b0;  *(v8bf*)&Bs[0][lofs + 8] = b1;
    for (int k0 = 0; k0 < K; k0 += 32) {
        const int cur = (k0 >> 5) & 1;
        __syncthreads();
        const bool more = (k0 + 32) < K;
        if (more) {
            a0 = *(const v8bf*)(gA + k0 + 32);  a1 = *(const v8bf*)(gA + k0 + 40);
            b0 = *(const v8bf*)(gB + k0 + 32);  b1 = *(const v8bf*)(gB + k0 + 40);
        }
        v16bf af[2], bfr[4];
#pragma unroll
        for (int tm = 0; tm < 2; ++tm)
            af[tm] = load_fragA(&As[cur][(wr * 32 + tm * 16) * 32], 32, lane);
#pragma unroll
        for (int tn = 0; tn < 4; ++tn)
            bfr[tn] = load_fragB(&Bs[cur][(wc * 64 + tn * 16) * 32], 32, lane);
#pragma unroll
        for (int tm = 0; tm < 2; ++tm)
#pragma unroll
            for (int tn = 0; tn < 4; ++tn)
                acc[tm][tn] = wmma_bf16(af[tm], bfr[tn], acc[tm][tn]);
        if (more) {
            const int nxt = cur ^ 1;
            *(v8bf*)&As[nxt][lofs] = a0;  *(v8bf*)&As[nxt][lofs + 8] = a1;
            *(v8bf*)&Bs[nxt][lofs] = b0;  *(v8bf*)&Bs[nxt][lofs + 8] = b1;
        }
    }
#endif

    // ---- epilogue (fully branch-free per template instantiation) ----
    const int mBase = m0 + wr * 32 + ((lane >> 4) << 3);
    const int nBase = n0 + wc * 64 + (lane & 15);
#pragma unroll
    for (int tm = 0; tm < 2; ++tm) {
#pragma unroll
        for (int tn = 0; tn < 4; ++tn) {
            const int n = nBase + tn * 16;
            if (!NG || n < Nreal) {
                float bv = BIAS ? bias[n] : 0.0f;
#pragma unroll
                for (int i = 0; i < 8; ++i) {
                    const int m = mBase + tm * 16 + i;
                    const size_t o = (size_t)m * ldo + n;
                    float t = acc[tm][tn][i];
                    if (BIAS)  t += bv;
                    if (GELU_) t = 0.5f * t * (1.0f + erff(t * 0.70710678118654752f));
                    if (RES)   t += resid[o];
                    if (STF)   outF[o] = t;
                    if (STB)   outB[o] = (bf16_t)t;
                }
            }
        }
    }
}

// ---------------- fused attention ----------------
// grid = B*H*2; each block: one head, 128 query rows.
// Dynamic LDS: Sf 128x256 f32 | Ab 128x256 bf16 | Vt 128x256 bf16 = 256 KB
#define ATTN_SMEM (128 * 256 * 4 + 128 * 256 * 2 + 128 * 256 * 2)

__global__ __launch_bounds__(256) void attn_kernel(
    const bf16_t* __restrict__ Q, const bf16_t* __restrict__ Km,
    const bf16_t* __restrict__ Vm, bf16_t* __restrict__ O, float scale)
{
    extern __shared__ char smem[];
    float*  Sf = (float*)smem;
    bf16_t* Ab = (bf16_t*)(smem + 128 * 256 * 4);
    bf16_t* Vt = (bf16_t*)(smem + 128 * 256 * 4 + 128 * 256 * 2);

    const int id   = blockIdx.x;
    const int rblk = id & 1;
    const int h    = (id >> 1) % H_;
    const int b    = id / (2 * H_);
    const int rowBase = rblk * 128;

    const int tid  = threadIdx.x;
    const int lane = tid & 31;
    const int wave = tid >> 5;
    const int wr   = wave >> 1;
    const int wc   = wave & 1;

    const size_t headOff = (size_t)h * D_;
    const bf16_t* Qh = Q  + (size_t)b * P_ * C_ + headOff;
    const bf16_t* Kh = Km + (size_t)b * P_ * C_ + headOff;
    const bf16_t* Vh = Vm + (size_t)b * P_ * C_ + headOff;

    // stage V transposed: Vt[d*256 + q]
    for (int e = tid; e < D_ * P_; e += 256) {
        int d = e >> 8;
        int q = e & 255;
        Vt[d * P_ + q] = Vh[(size_t)q * C_ + d];
    }

    // ---- phase 1: S = Q K^T (128 x 256) ----
    const bf16_t* qrow = Qh + (size_t)(rowBase + wr * 32) * C_;
    for (int cc = 0; cc < 2; ++cc) {
        const int c0 = wc * 128 + cc * 64;
        v8f acc[2][4];
#pragma unroll
        for (int tm = 0; tm < 2; ++tm)
#pragma unroll
            for (int tn = 0; tn < 4; ++tn) acc[tm][tn] = zero8();

#pragma unroll
        for (int k0 = 0; k0 < D_; k0 += 32) {
            v16bf af[2], bfr[4];
#pragma unroll
            for (int tm = 0; tm < 2; ++tm)
                af[tm] = load_fragA(qrow + (size_t)tm * 16 * C_ + k0, C_, lane);
#pragma unroll
            for (int tn = 0; tn < 4; ++tn)
                bfr[tn] = load_fragB(Kh + (size_t)(c0 + tn * 16) * C_ + k0, C_, lane);
#pragma unroll
            for (int tm = 0; tm < 2; ++tm)
#pragma unroll
                for (int tn = 0; tn < 4; ++tn)
                    acc[tm][tn] = wmma_bf16(af[tm], bfr[tn], acc[tm][tn]);
        }
        const int rB = wr * 32 + ((lane >> 4) << 3);
        const int nB = lane & 15;
#pragma unroll
        for (int tm = 0; tm < 2; ++tm)
#pragma unroll
            for (int tn = 0; tn < 4; ++tn)
#pragma unroll
                for (int i = 0; i < 8; ++i)
                    Sf[(rB + tm * 16 + i) * 256 + c0 + tn * 16 + nB] = acc[tm][tn][i];
    }
    __syncthreads();

    // ---- phase 2: causal softmax; 16 rows per wave ----
    for (int rr = 0; rr < 16; ++rr) {
        const int r = wave * 16 + rr;
        const int p = rowBase + r;
        float vals[8];
        float mx = -3.0e38f;
#pragma unroll
        for (int j = 0; j < 8; ++j) {
            int c = lane + j * 32;
            float v = (c <= p) ? Sf[r * 256 + c] * scale : -3.0e38f;
            vals[j] = v;
            mx = fmaxf(mx, v);
        }
        mx = wave_max(mx);
        float sum = 0.0f;
#pragma unroll
        for (int j = 0; j < 8; ++j) { vals[j] = expf(vals[j] - mx); sum += vals[j]; }
        sum = wave_sum(sum);
        float inv = 1.0f / sum;
#pragma unroll
        for (int j = 0; j < 8; ++j)
            Ab[r * 256 + lane + j * 32] = (bf16_t)(vals[j] * inv);
    }
    __syncthreads();

    // ---- phase 3: O = softmax(S) * V  (128 x 128, K=256) ----
    {
        v8f acc[2][4];
#pragma unroll
        for (int tm = 0; tm < 2; ++tm)
#pragma unroll
            for (int tn = 0; tn < 4; ++tn) acc[tm][tn] = zero8();

        for (int k0 = 0; k0 < P_; k0 += 32) {
            v16bf af[2], bfr[4];
#pragma unroll
            for (int tm = 0; tm < 2; ++tm)
                af[tm] = load_fragA(&Ab[(wr * 32 + tm * 16) * 256 + k0], 256, lane);
#pragma unroll
            for (int tn = 0; tn < 4; ++tn)
                bfr[tn] = load_fragB(&Vt[(wc * 64 + tn * 16) * 256 + k0], 256, lane);
#pragma unroll
            for (int tm = 0; tm < 2; ++tm)
#pragma unroll
                for (int tn = 0; tn < 4; ++tn)
                    acc[tm][tn] = wmma_bf16(af[tm], bfr[tn], acc[tm][tn]);
        }
        const int mB = rowBase + wr * 32 + ((lane >> 4) << 3);
        const int nB = wc * 64 + (lane & 15);
#pragma unroll
        for (int tm = 0; tm < 2; ++tm)
#pragma unroll
            for (int tn = 0; tn < 4; ++tn)
#pragma unroll
                for (int i = 0; i < 8; ++i) {
                    int m = mB + tm * 16 + i;
                    int n = nB + tn * 16;
                    O[((size_t)(b * P_ + m)) * C_ + headOff + n] = (bf16_t)acc[tm][tn][i];
                }
    }
}

// ---------------- launcher ----------------
extern "C" void kernel_launch(void* const* d_in, const int* in_sizes, int n_in,
                              void* d_out, int out_size, void* d_ws, size_t ws_size,
                              hipStream_t stream) {
    (void)in_sizes; (void)n_in; (void)out_size; (void)ws_size;
    const int*   idx     = (const int*)  d_in[0];
    const float* tok_emb = (const float*)d_in[1];
    const float* pos_emb = (const float*)d_in[2];
    const float* wq      = (const float*)d_in[3];
    const float* wk      = (const float*)d_in[4];
    const float* wv      = (const float*)d_in[5];
    const float* attn_pw = (const float*)d_in[6];
    const float* attn_pb = (const float*)d_in[7];
    const float* mlp_w1  = (const float*)d_in[8];
    const float* mlp_b1  = (const float*)d_in[9];
    const float* mlp_w2  = (const float*)d_in[10];
    const float* mlp_b2  = (const float*)d_in[11];
    const float* pre_g   = (const float*)d_in[12];
    const float* pre_b   = (const float*)d_in[13];
    const float* post_g  = (const float*)d_in[14];
    const float* post_b  = (const float*)d_in[15];
    const float* lnf_g   = (const float*)d_in[16];
    const float* lnf_b   = (const float*)d_in[17];
    const float* lm_w    = (const float*)d_in[18];
    const float* lm_b    = (const float*)d_in[19];

    char* ws = (char*)d_ws;
    size_t off = 0;
    auto alloc = [&](size_t bytes) -> char* {
        char* p = ws + off;
        off += (bytes + 255) & ~(size_t)255;
        return p;
    };

    float*  x   = (float*) alloc((size_t)M_ * C_ * 4);
    float*  x2  = (float*) alloc((size_t)M_ * C_ * 4);
    bf16_t* hb  = (bf16_t*)alloc((size_t)M_ * C_ * 2);
    bf16_t* qb  = (bf16_t*)alloc((size_t)M_ * C_ * 2);
    bf16_t* kb  = (bf16_t*)alloc((size_t)M_ * C_ * 2);
    bf16_t* vb  = (bf16_t*)alloc((size_t)M_ * C_ * 2);
    bf16_t* ob  = (bf16_t*)alloc((size_t)M_ * C_ * 2);
    bf16_t* m1  = (bf16_t*)alloc((size_t)M_ * FF_ * 2);
    bf16_t* wqt = (bf16_t*)alloc((size_t)C_ * C_ * 2);
    bf16_t* wkt = (bf16_t*)alloc((size_t)C_ * C_ * 2);
    bf16_t* wvt = (bf16_t*)alloc((size_t)C_ * C_ * 2);
    bf16_t* pwt = (bf16_t*)alloc((size_t)C_ * C_ * 2);
    bf16_t* w1t = (bf16_t*)alloc((size_t)FF_ * C_ * 2);
    bf16_t* w2t = (bf16_t*)alloc((size_t)C_ * FF_ * 2);
    bf16_t* lmt = (bf16_t*)alloc((size_t)128 * C_ * 2);

    const dim3 blk(256);

    embed_kernel<<<(M_ * C_) / 256, blk, 0, stream>>>(idx, tok_emb, pos_emb, x);
    head_transpose<<<(C_ * C_ + 255) / 256, blk, 0, stream>>>(wq, wqt);
    head_transpose<<<(C_ * C_ + 255) / 256, blk, 0, stream>>>(wk, wkt);
    head_transpose<<<(C_ * C_ + 255) / 256, blk, 0, stream>>>(wv, wvt);
    transpose_conv<<<(C_ * C_ + 255) / 256, blk, 0, stream>>>(attn_pw, pwt, C_, C_, C_);
    transpose_conv<<<(FF_ * C_ + 255) / 256, blk, 0, stream>>>(mlp_w1, w1t, C_, FF_, FF_);
    transpose_conv<<<(C_ * FF_ + 255) / 256, blk, 0, stream>>>(mlp_w2, w2t, FF_, C_, C_);
    transpose_conv<<<(128 * C_ + 255) / 256, blk, 0, stream>>>(lm_w, lmt, C_, V_, 128);

    // pre-LN -> QKV (bf16 out only)
    ln_kernel<<<M_ / 8, blk, 0, stream>>>(x, pre_g, pre_b, hb, M_, C_);
    dim3 gC(C_ / 128, M_ / 128);
    gemm_kernel<false,false,false,false,true,false><<<gC, blk, 0, stream>>>(
        hb, wqt, nullptr, nullptr, nullptr, qb, C_, C_, C_);
    gemm_kernel<false,false,false,false,true,false><<<gC, blk, 0, stream>>>(
        hb, wkt, nullptr, nullptr, nullptr, kb, C_, C_, C_);
    gemm_kernel<false,false,false,false,true,false><<<gC, blk, 0, stream>>>(
        hb, wvt, nullptr, nullptr, nullptr, vb, C_, C_, C_);

    // attention (reference scales by dim_embd^-0.5 = C^-0.5)
    const float scale = 1.0f / sqrtf((float)C_);
    attn_kernel<<<B_ * H_ * 2, blk, ATTN_SMEM, stream>>>(qb, kb, vb, ob, scale);

    // projection: bias + residual(x) -> x2 (f32)
    gemm_kernel<true,true,false,true,false,false><<<gC, blk, 0, stream>>>(
        ob, pwt, attn_pb, x, x2, nullptr, C_, C_, C_);

    // MLP
    ln_kernel<<<M_ / 8, blk, 0, stream>>>(x2, post_g, post_b, hb, M_, C_);
    dim3 gF(FF_ / 128, M_ / 128);
    gemm_kernel<true,false,true,false,true,false><<<gF, blk, 0, stream>>>(
        hb, w1t, mlp_b1, nullptr, nullptr, m1, C_, FF_, FF_);
    gemm_kernel<true,true,false,true,false,false><<<gC, blk, 0, stream>>>(
        m1, w2t, mlp_b2, x2, x, nullptr, FF_, C_, C_);

    // final LN + LM head (N padded 65 -> 128, stores guarded to 65)
    ln_kernel<<<M_ / 8, blk, 0, stream>>>(x, lnf_g, lnf_b, hb, M_, C_);
    dim3 gV(1, M_ / 128);
    gemm_kernel<true,false,false,true,false,true><<<gV, blk, 0, stream>>>(
        hb, lmt, lm_b, nullptr, (float*)d_out, nullptr, C_, V_, V_);
}
